// TttM2BMMModule_51754355917057
// MI455X (gfx1250) — matrix operand order repaired and checked by reading, not compiled, not run on hardware
//
#include <hip/hip_runtime.h>

typedef __attribute__((ext_vector_type(16))) __bf16 v16bf;
typedef __attribute__((ext_vector_type(8)))  float  v8f;
typedef __attribute__((ext_vector_type(4)))  unsigned int u32x4;
typedef __attribute__((ext_vector_type(8)))  int i32x8;
typedef __attribute__((ext_vector_type(4)))  int i32x4;

// Problem constants
constexpr int Bc  = 4;
constexpr int Lc  = 4096;
constexpr int Cc  = 1024;
constexpr int NHc = 16;
constexpr int CSc = 16;
constexpr int HFc = 64;     // C / NH
constexpr int HF4 = 256;    // 4 * HF
constexpr int NCc = 256;    // L / CS

// ---------------------------------------------------------------------------
// WMMA helper: D = A(16x32 bf16) * B(32x16 bf16) + C(16x16 f32)
// ---------------------------------------------------------------------------
__device__ __forceinline__ v8f wmma_bf16(v16bf a, v16bf b, v8f c) {
  return __builtin_amdgcn_wmma_f32_16x16x32_bf16(false, a, false, b, (short)0, c,
                                                 false, false);
}

// ---------------------------------------------------------------------------
// Tensor Data Mover: 2D bf16 tile load, global -> LDS (ISA 08_async_tensor §8)
//   group0: [1:0]=count=1, [63:32]=lds_addr, [120:64]=global_addr, [127:126]=2
//   group1: [17:16]=data_size(1=2B), [79:48]=tensor_dim0, [111:80]=tensor_dim1,
//           [127:112]=tile_dim0, [143:128]=tile_dim1, [207:160]=dim0_stride
// ---------------------------------------------------------------------------
__device__ __forceinline__ unsigned lds_off32(const void* p) {
  // generic pointers into LDS carry the LDS byte offset in the low 32 bits
  return (unsigned)(unsigned long long)p;
}

__device__ __forceinline__ void tdm_load_2d_bf16(unsigned lds_byte,
                                                 const void* gptr,
                                                 unsigned tdim0, unsigned tdim1,
                                                 unsigned tile0, unsigned tile1,
                                                 unsigned stride0) {
  unsigned long long ga = (unsigned long long)gptr;
  u32x4 g0;
  g0[0] = 1u;                                            // count = 1 descriptor
  g0[1] = lds_byte;                                      // lds_addr
  g0[2] = (unsigned)ga;                                  // global_addr[31:0]
  g0[3] = (unsigned)((ga >> 32) & 0x01FFFFFFu) | (2u << 30);  // [56:32] | type=2
  i32x8 g1;
  g1[0] = (int)(1u << 16);                               // data_size = 2 bytes
  g1[1] = (int)(tdim0 << 16);                            // tensor_dim0[15:0]
  g1[2] = (int)((tdim0 >> 16) & 0xFFFFu) | (int)(tdim1 << 16);
  g1[3] = (int)((tdim1 >> 16) & 0xFFFFu) | (int)(tile0 << 16);
  g1[4] = (int)(tile1 & 0xFFFFu);                        // tile_dim1 (tile_dim2=0)
  g1[5] = (int)stride0;                                  // dim0_stride[31:0]
  g1[6] = 0;                                             // dim0_stride[47:32]
  g1[7] = 0;
  i32x4 z4 = {0, 0, 0, 0};
#if defined(__clang_major__) && (__clang_major__ >= 23)
  i32x8 z8 = {0, 0, 0, 0, 0, 0, 0, 0};
  __builtin_amdgcn_tensor_load_to_lds(g0, g1, z4, z4, z8, 0);
#else
  __builtin_amdgcn_tensor_load_to_lds(g0, g1, z4, z4, 0);
#endif
}

// s_wait_tensorcnt takes an immediate -> compile-time constant template arg
template <int N>
__device__ __forceinline__ void wait_tensorcnt() {
  __builtin_amdgcn_s_wait_tensorcnt(N);
}

// ---------------------------------------------------------------------------
// Fragment loaders (ISA 7.12.2 layouts)
// A 16-bit 16x32: lane m = l&15, hi = l>>4; v<4 -> K=8*hi+2v, v>=4 -> K=16+8*hi+2(v-4)
// B 32x16: same K order as A; v<4 -> K=8*hi+2v, v>=4 -> K=16+8*hi+2(v-4)
// Acc 16x16 f32: M = r + 8*(lane>>4), N = lane&15
// ---------------------------------------------------------------------------
__device__ __forceinline__ int a_kidx(int v, int hi) {
  return (v < 4) ? (8 * hi + 2 * v) : (16 + 8 * hi + 2 * (v - 4));
}
__device__ __forceinline__ int b_kidx(int v, int hi) { return a_kidx(v, hi); }

__device__ __forceinline__ v16bf fragA_rm(const __bf16* base, int row0, int col0,
                                          int ld) {
  const int lane = threadIdx.x & 31;
  const int m = lane & 15, hi = lane >> 4;
  v16bf f;
#pragma unroll
  for (int v = 0; v < 8; ++v) {
    int k = a_kidx(v, hi);
    const __bf16* p = base + (row0 + m) * ld + col0 + k;
    f[2 * v]     = p[0];
    f[2 * v + 1] = p[1];
  }
  return f;
}

__device__ __forceinline__ v16bf fragA_k16(const __bf16* base) {
  const int lane = threadIdx.x & 31;
  const int m = lane & 15, hi = lane >> 4;
  v16bf f;
#pragma unroll
  for (int v = 0; v < 8; ++v) {
    int k = a_kidx(v, hi);
    f[2 * v]     = (k < 16)     ? base[m * 16 + k]     : (__bf16)0.0f;
    f[2 * v + 1] = (k + 1 < 16) ? base[m * 16 + k + 1] : (__bf16)0.0f;
  }
  return f;
}

// A = (s * S)^T, S is 16 x ld row-major; A[i][k] = s * S[k][i], K padded 16->32
__device__ __forceinline__ v16bf fragA_scaleT_k16(const __bf16* src, int row0,
                                                  int ld, float s) {
  const int lane = threadIdx.x & 31;
  const int m = lane & 15, hi = lane >> 4;
  v16bf f;
#pragma unroll
  for (int v = 0; v < 8; ++v) {
    int k = a_kidx(v, hi);
    f[2 * v] = (k < 16) ? (__bf16)(s * (float)src[k * ld + row0 + m]) : (__bf16)0.0f;
    f[2 * v + 1] =
        (k + 1 < 16) ? (__bf16)(s * (float)src[(k + 1) * ld + row0 + m]) : (__bf16)0.0f;
  }
  return f;
}

// B from "BT" storage: base[n * ldk + k] == B[k][n]  (K contiguous)
__device__ __forceinline__ v16bf fragB_bt(const __bf16* base, int k0, int n0,
                                          int ldk) {
  const int lane = threadIdx.x & 31;
  const int n = lane & 15, hi = lane >> 4;
  v16bf f;
#pragma unroll
  for (int v = 0; v < 8; ++v) {
    int k = b_kidx(v, hi);
    const __bf16* p = base + (n0 + n) * ldk + k0 + k;
    f[2 * v]     = p[0];
    f[2 * v + 1] = p[1];
  }
  return f;
}

// B from plain row-major storage: base[k * ldn + n] == B[k][n]
__device__ __forceinline__ v16bf fragB_rm(const __bf16* base, int k0, int n0,
                                          int ldn) {
  const int lane = threadIdx.x & 31;
  const int n = lane & 15, hi = lane >> 4;
  v16bf f;
#pragma unroll
  for (int v = 0; v < 8; ++v) {
    int k = b_kidx(v, hi);
    f[2 * v]     = base[(k0 + k) * ldn + n0 + n];
    f[2 * v + 1] = base[(k0 + k + 1) * ldn + n0 + n];
  }
  return f;
}

// B, 16 real K rows (row-major, leading dim ldn), zero padded to 32
__device__ __forceinline__ v16bf fragB_k16(const __bf16* base, int n0, int ldn) {
  const int lane = threadIdx.x & 31;
  const int n = lane & 15, hi = lane >> 4;
  v16bf f;
#pragma unroll
  for (int v = 0; v < 8; ++v) {
    int k = b_kidx(v, hi);
    f[2 * v]     = (k < 16)     ? base[k * ldn + n0 + n]       : (__bf16)0.0f;
    f[2 * v + 1] = (k + 1 < 16) ? base[(k + 1) * ldn + n0 + n] : (__bf16)0.0f;
  }
  return f;
}

// ---------------------------------------------------------------------------
// f32 -> bf16 converter
// ---------------------------------------------------------------------------
__global__ void cvt_f32_bf16(const float* __restrict__ in, __bf16* __restrict__ out,
                             int n) {
  int i = blockIdx.x * blockDim.x + threadIdx.x;
  if (i < n) out[i] = (__bf16)in[i];
}

// ---------------------------------------------------------------------------
// ilr = sigmoid(hs @ ilr_w + ilr_b), stored as (B, NH, NC, CS) f32
// ---------------------------------------------------------------------------
__global__ void ilr_kernel(const float* __restrict__ hs, const float* __restrict__ w,
                           const float* __restrict__ bias, float* __restrict__ out) {
  int idx = blockIdx.x * blockDim.x + threadIdx.x;  // B*L*NH
  int h = idx & (NHc - 1);
  int l = (idx >> 4) & (Lc - 1);
  int b = idx >> 16;
  const float* hp = hs + ((size_t)b * Lc + l) * Cc;
  float s = bias[h];
  for (int k = 0; k < Cc; ++k) s += hp[k] * w[k * NHc + h];
  float sig = 1.0f / (1.0f + __expf(-s));
  int ck = l >> 4, m = l & 15;
  out[(((size_t)b * NHc + h) * NCc + ck) * CSc + m] = sig;
}

// ---------------------------------------------------------------------------
// Generic bf16 GEMM: C(MxN) = A(MxK) * B(KxN), A/B bf16 row-major.
// 128x128 block tile, BK=32, 8 waves (2x4 WMMA tiles each).
// Staging: TDM tensor_load_to_lds, double buffered; wave 0 issues DMAs and
// gates consumption with s_wait_tensorcnt (in-order completion).
// ---------------------------------------------------------------------------
template <bool BF16OUT>
__global__ __launch_bounds__(256) void gemm_bf16_kernel(const __bf16* __restrict__ A,
                                                        const __bf16* __restrict__ Bm,
                                                        void* __restrict__ Cp, int M,
                                                        int N, int K) {
  __shared__ __align__(16) __bf16 As[2][128 * 32];  // row-major 128 x 32
  __shared__ __align__(16) __bf16 Bs[2][32 * 128];  // row-major 32 x 128
  const int tid = threadIdx.x, lane = tid & 31, wave = tid >> 5;
  const int accM0 = 8 * (lane >> 4), accN = lane & 15;
  const int rowBase = blockIdx.y * 128;
  const int colBase = blockIdx.x * 128;
  const int mt0 = 2 * (wave & 3);
  const int nt0 = 4 * (wave >> 2);
  v8f acc[2][4] = {};

  auto issue = [&](int k0, int buf) {
    // A tile: 128 rows x 32 cols, row stride K
    tdm_load_2d_bf16(lds_off32(&As[buf][0]), A + (size_t)rowBase * K + k0,
                     (unsigned)K, (unsigned)M, 32u, 128u, (unsigned)K);
    // B tile: 32 rows x 128 cols, row stride N
    tdm_load_2d_bf16(lds_off32(&Bs[buf][0]), Bm + (size_t)k0 * N + colBase,
                     (unsigned)N, (unsigned)K, 128u, 32u, (unsigned)N);
  };

  if (wave == 0) {
    issue(0, 0);
    if (K > 32) issue(32, 1);
  }

  for (int k0 = 0; k0 < K; k0 += 32) {
    const int buf = (k0 >> 5) & 1;
    if (wave == 0) {
      if (k0 + 32 < K) wait_tensorcnt<2>();
      else             wait_tensorcnt<0>();
    }
    __syncthreads();  // data for `buf` landed (wave0 gated it)

    v16bf aF[2], bF[4];
#pragma unroll
    for (int i = 0; i < 2; ++i) aF[i] = fragA_rm(&As[buf][0], 16 * (mt0 + i), 0, 32);
#pragma unroll
    for (int q = 0; q < 4; ++q) bF[q] = fragB_rm(&Bs[buf][0], 0, 16 * (nt0 + q), 128);
#pragma unroll
    for (int i = 0; i < 2; ++i)
#pragma unroll
      for (int q = 0; q < 4; ++q) acc[i][q] = wmma_bf16(aF[i], bF[q], acc[i][q]);

    __syncthreads();  // all waves done with `buf` before the DMA refills it
    if (wave == 0 && k0 + 64 < K) issue(k0 + 64, buf);
  }

#pragma unroll
  for (int i = 0; i < 2; ++i)
#pragma unroll
    for (int q = 0; q < 4; ++q) {
      int gr0 = rowBase + 16 * (mt0 + i) + accM0;
      int gc = colBase + 16 * (nt0 + q) + accN;
#pragma unroll
      for (int r = 0; r < 8; ++r) {
        if (BF16OUT)
          ((__bf16*)Cp)[(size_t)(gr0 + r) * N + gc] = (__bf16)acc[i][q][r];
        else
          ((float*)Cp)[(size_t)(gr0 + r) * N + gc] = acc[i][q][r];
      }
    }
}

// ---------------------------------------------------------------------------
// TTT scan kernel.  One block per (b, h).  Fast weights W1 (64x256) and
// W2 (256x64) live as f32 WMMA accumulators (8 tiles per wave each), with bf16
// operand mirrors in LDS:
//   W1bT[n*64  + k] = W1[k][n]   (BT layout for Z1/Z1_bar)
//   W2bT[n*256 + k] = W2[k][n]   (BT layout for Z2/Z2_bar; row-major W2^T for g1)
// coeff / tril / sign factors folded into A operands.  XB/XC chunk tiles are
// prefetched one chunk ahead by the TDM into double buffers.
// ---------------------------------------------------------------------------
__global__ __launch_bounds__(256) void ttt_scan_kernel(
    const __bf16* __restrict__ XAg, const __bf16* __restrict__ XBg,
    const __bf16* __restrict__ XCg, const float* __restrict__ ilr_g,
    const float* __restrict__ W1g, const float* __restrict__ W2g,
    __bf16* __restrict__ Yg) {
  __shared__ __align__(16) __bf16 W1bT[HF4 * HFc];  // 256x64
  __shared__ __align__(16) __bf16 W2bT[HFc * HF4];  // 64x256
  __shared__ __align__(16) __bf16 XBb[2][CSc * HFc];  // 16x64 row-major, dbl-buf
  __shared__ __align__(16) __bf16 XCb[2][CSc * HFc];
  __shared__ __align__(16) __bf16 Z1b[CSc * HF4];     // 16x256 row-major
  __shared__ __align__(16) __bf16 Z1barb[CSc * HF4];  // 16x256
  __shared__ __align__(16) __bf16 g1b[CSc * HF4];     // 16x256
  __shared__ __align__(16) __bf16 g2b[CSc * HFc];     // 16x64
  __shared__ __align__(16) __bf16 A1s[CSc * CSc];     // -coeff * tril(Attn1)
  __shared__ __align__(16) __bf16 A2s[CSc * CSc];     // -coeff * tril(Attn2)
  __shared__ float coeff[CSc];

  const int tid = threadIdx.x, lane = tid & 31, wave = tid >> 5;
  const int bh = blockIdx.x;
  const int b = bh >> 4, h = bh & (NHc - 1);
  const int accM0 = 8 * (lane >> 4), accN = lane & 15;

  auto issue_chunk = [&](int ck) {
    const size_t xb = ((size_t)b * Lc + (size_t)ck * CSc) * Cc + (size_t)h * HFc;
    const int buf = ck & 1;
    tdm_load_2d_bf16(lds_off32(&XBb[buf][0]), XBg + xb, (unsigned)Cc,
                     (unsigned)(Bc * Lc), (unsigned)HFc, (unsigned)CSc,
                     (unsigned)Cc);
    tdm_load_2d_bf16(lds_off32(&XCb[buf][0]), XCg + xb, (unsigned)Cc,
                     (unsigned)(Bc * Lc), (unsigned)HFc, (unsigned)CSc,
                     (unsigned)Cc);
  };

  // ---- init persistent fast-weight accumulators + LDS mirrors ----
  if (wave == 0) issue_chunk(0);  // prefetch chunk 0 while we set up

  v8f w1acc[8], w2acc[8];
  const float* W1h = W1g + (size_t)h * HFc * HF4;
  const float* W2h = W2g + (size_t)h * HF4 * HFc;
  const int mt1 = wave & 3;             // W1 tile row owned by this wave
  const int nt1base = 8 * (wave >> 2);  // W1 tile cols: nt1base .. +7
#pragma unroll
  for (int j = 0; j < 8; ++j) {
    int nt = nt1base + j;
#pragma unroll
    for (int r = 0; r < 8; ++r)
      w1acc[j][r] = W1h[(size_t)(16 * mt1 + accM0 + r) * HF4 + 16 * nt + accN];
  }
#pragma unroll
  for (int j = 0; j < 8; ++j) {
    int mt = 2 * wave + (j >> 2), nt = j & 3;
#pragma unroll
    for (int r = 0; r < 8; ++r)
      w2acc[j][r] = W2h[(size_t)(16 * mt + accM0 + r) * HFc + 16 * nt + accN];
  }
  for (int i = tid; i < HFc * HF4; i += 256) {
    int k = i >> 8, n = i & 255;  // W1: k<64, n<256
    W1bT[n * 64 + k] = (__bf16)W1h[(size_t)k * HF4 + n];
  }
  for (int i = tid; i < HF4 * HFc; i += 256) {
    int k = i >> 6, n = i & 63;  // W2: k<256, n<64
    W2bT[n * 256 + k] = (__bf16)W2h[(size_t)k * HFc + n];
  }
  __syncthreads();

  for (int ck = 0; ck < NCc; ++ck) {
    const size_t xbase = ((size_t)b * Lc + (size_t)ck * CSc) * Cc + (size_t)h * HFc;
    const __bf16* XBc = &XBb[ck & 1][0];
    const __bf16* XCc = &XCb[ck & 1][0];

    // ---- s0: prefetch next chunk, gate current chunk's tiles, coeff ----
    if (wave == 0) {
      if (ck + 1 < NCc) {
        issue_chunk(ck + 1);
        wait_tensorcnt<2>();
      } else {
        wait_tensorcnt<0>();
      }
    }
    if (tid < CSc) {
      float iv = ilr_g[(((size_t)b * NHc + h) * NCc + ck) * CSc + tid];
      coeff[tid] = (1.0f / (float)(tid + 1)) * iv / (float)HFc;  // LR = 1
    }
    __syncthreads();

    // ---- s1: Z1 = XB @ W1  (16x256), 2 N-tiles per wave ----
#pragma unroll
    for (int q = 0; q < 2; ++q) {
      int nt = 2 * wave + q;
      v8f acc = {};
      acc = wmma_bf16(fragA_rm(XBc, 0, 0, 64), fragB_bt(W1bT, 0, 16 * nt, 64), acc);
      acc = wmma_bf16(fragA_rm(XBc, 0, 32, 64), fragB_bt(W1bT, 32, 16 * nt, 64), acc);
#pragma unroll
      for (int r = 0; r < 8; ++r)
        Z1b[(accM0 + r) * 256 + 16 * nt + accN] = (__bf16)acc[r];
    }
    __syncthreads();

    // ---- s2: Z2 = Z1 @ W2, g2 = Z2 - XA (waves 0-3); Attn1 (wave 4) ----
    if (wave < 4) {
      int nt = wave;
      v8f acc = {};
#pragma unroll
      for (int kk = 0; kk < 8; ++kk)
        acc = wmma_bf16(fragA_rm(Z1b, 0, 32 * kk, 256),
                        fragB_bt(W2bT, 32 * kk, 16 * nt, 256), acc);
#pragma unroll
      for (int r = 0; r < 8; ++r) {
        float xa = (float)XAg[xbase + (size_t)(accM0 + r) * Cc + 16 * nt + accN];
        g2b[(accM0 + r) * 64 + 16 * nt + accN] = (__bf16)(acc[r] - xa);
      }
    } else if (wave == 4) {
      // Attn1 = tril(XC @ XB^T); store A1s = -coeff_m * Attn1
      v8f acc = {};
      acc = wmma_bf16(fragA_rm(XCc, 0, 0, 64), fragB_bt(XBc, 0, 0, 64), acc);
      acc = wmma_bf16(fragA_rm(XCc, 0, 32, 64), fragB_bt(XBc, 32, 0, 64), acc);
#pragma unroll
      for (int r = 0; r < 8; ++r) {
        int M = accM0 + r;
        float v = (accN <= M) ? acc[r] : 0.0f;
        A1s[M * 16 + accN] = (__bf16)(-coeff[M] * v);
      }
    }
    __syncthreads();

    // ---- s3: g1 = g2 @ W2^T  (W2bT read row-major == W2^T row-major) ----
#pragma unroll
    for (int q = 0; q < 2; ++q) {
      int nt = 2 * wave + q;
      v8f acc = {};
      acc = wmma_bf16(fragA_rm(g2b, 0, 0, 64), fragB_rm(W2bT, 0, 16 * nt, 256), acc);
      acc = wmma_bf16(fragA_rm(g2b, 0, 32, 64), fragB_rm(W2bT, 32, 16 * nt, 256), acc);
#pragma unroll
      for (int r = 0; r < 8; ++r)
        g1b[(accM0 + r) * 256 + 16 * nt + accN] = (__bf16)acc[r];
    }
    __syncthreads();

    // ---- s4: Z1_bar = XC @ W1 + A1s @ g1(pad K=16) ----
#pragma unroll
    for (int q = 0; q < 2; ++q) {
      int nt = 2 * wave + q;
      v8f acc = {};
      acc = wmma_bf16(fragA_rm(XCc, 0, 0, 64), fragB_bt(W1bT, 0, 16 * nt, 64), acc);
      acc = wmma_bf16(fragA_rm(XCc, 0, 32, 64), fragB_bt(W1bT, 32, 16 * nt, 64), acc);
      acc = wmma_bf16(fragA_k16(A1s), fragB_k16(g1b, 16 * nt, 256), acc);
#pragma unroll
      for (int r = 0; r < 8; ++r)
        Z1barb[(accM0 + r) * 256 + 16 * nt + accN] = (__bf16)acc[r];
    }
    __syncthreads();

    // ---- s5: Attn2 (wave 7); W1 update + commit; W2 update (regs only) ----
    const float negCl = -coeff[15];  // -coeff_last
    if (wave == 7) {
      v8f acc = {};
#pragma unroll
      for (int kk = 0; kk < 8; ++kk)
        acc = wmma_bf16(fragA_rm(Z1barb, 0, 32 * kk, 256),
                        fragB_bt(Z1b, 32 * kk, 0, 256), acc);
#pragma unroll
      for (int r = 0; r < 8; ++r) {
        int M = accM0 + r;
        float v = (accN <= M) ? acc[r] : 0.0f;
        A2s[M * 16 + accN] = (__bf16)(-coeff[M] * v);
      }
    }
    {  // W1 += (-coeff_last * XB)^T @ g1 ; commit bf16 mirror
      v16bf aT = fragA_scaleT_k16(XBc, 16 * mt1, 64, negCl);
#pragma unroll
      for (int j = 0; j < 8; ++j) {
        int nt = nt1base + j;
        w1acc[j] = wmma_bf16(aT, fragB_k16(g1b, 16 * nt, 256), w1acc[j]);
#pragma unroll
        for (int r = 0; r < 8; ++r)
          W1bT[(16 * nt + accN) * 64 + 16 * mt1 + accM0 + r] = (__bf16)w1acc[j][r];
      }
    }
    {  // W2 += (-coeff_last * Z1)^T @ g2  (commit deferred past Z2_bar reads)
      v16bf aT0 = fragA_scaleT_k16(Z1b, 16 * (2 * wave), 256, negCl);
      v16bf aT1 = fragA_scaleT_k16(Z1b, 16 * (2 * wave + 1), 256, negCl);
#pragma unroll
      for (int j = 0; j < 8; ++j) {
        int nt = j & 3;
        w2acc[j] = wmma_bf16((j >> 2) ? aT1 : aT0, fragB_k16(g2b, 16 * nt, 64),
                             w2acc[j]);
      }
    }
    __syncthreads();

    // ---- s6: Z2_bar = Z1_bar @ W2_old + A2s @ g2(pad) -> output ----
    if (wave < 4) {
      int nt = wave;
      v8f acc = {};
#pragma unroll
      for (int kk = 0; kk < 8; ++kk)
        acc = wmma_bf16(fragA_rm(Z1barb, 0, 32 * kk, 256),
                        fragB_bt(W2bT, 32 * kk, 16 * nt, 256), acc);
      acc = wmma_bf16(fragA_k16(A2s), fragB_k16(g2b, 16 * nt, 64), acc);
#pragma unroll
      for (int r = 0; r < 8; ++r)
        Yg[xbase + (size_t)(accM0 + r) * Cc + 16 * nt + accN] = (__bf16)acc[r];
    }
    __syncthreads();

    // ---- s6b: commit W2 bf16 mirror ----
#pragma unroll
    for (int j = 0; j < 8; ++j) {
      int mt = 2 * wave + (j >> 2), nt = j & 3;
#pragma unroll
      for (int r = 0; r < 8; ++r)
        W2bT[(16 * nt + accN) * 256 + 16 * mt + accM0 + r] = (__bf16)w2acc[j][r];
    }
    __syncthreads();
  }
}

// ---------------------------------------------------------------------------
extern "C" void kernel_launch(void* const* d_in, const int* in_sizes, int n_in,
                              void* d_out, int out_size, void* d_ws, size_t ws_size,
                              hipStream_t stream) {
  const float* hs    = (const float*)d_in[0];
  const float* Wq    = (const float*)d_in[1];
  const float* Wk    = (const float*)d_in[2];
  const float* Wv    = (const float*)d_in[3];
  const float* Wo    = (const float*)d_in[4];
  const float* ilr_w = (const float*)d_in[5];
  const float* ilr_b = (const float*)d_in[6];
  const float* W1g   = (const float*)d_in[7];
  const float* W2g   = (const float*)d_in[8];

  const int HS_N = Bc * Lc * Cc;  // 16,777,216
  const int W_N  = Cc * Cc;       // 1,048,576
  const int ILR_N = Bc * NHc * NCc * CSc;

  char* ws = (char*)d_ws;
  size_t off = 0;
  auto take = [&](size_t bytes) {
    char* p = ws + off;
    off += (bytes + 255) & ~(size_t)255;
    return p;
  };
  __bf16* hs_bf = (__bf16*)take((size_t)HS_N * 2);
  __bf16* wq_bf = (__bf16*)take((size_t)W_N * 2);
  __bf16* wk_bf = (__bf16*)take((size_t)W_N * 2);
  __bf16* wv_bf = (__bf16*)take((size_t)W_N * 2);
  __bf16* wo_bf = (__bf16*)take((size_t)W_N * 2);
  __bf16* xc_bf = (__bf16*)take((size_t)HS_N * 2);
  __bf16* xb_bf = (__bf16*)take((size_t)HS_N * 2);
  __bf16* xa_bf = (__bf16*)take((size_t)HS_N * 2);
  __bf16* y_bf  = (__bf16*)take((size_t)HS_N * 2);
  float*  ilr_f = (float*)take((size_t)ILR_N * 4);

  // 1) convert operands to bf16
  cvt_f32_bf16<<<(HS_N + 255) / 256, 256, 0, stream>>>(hs, hs_bf, HS_N);
  cvt_f32_bf16<<<(W_N + 255) / 256, 256, 0, stream>>>(Wq, wq_bf, W_N);
  cvt_f32_bf16<<<(W_N + 255) / 256, 256, 0, stream>>>(Wk, wk_bf, W_N);
  cvt_f32_bf16<<<(W_N + 255) / 256, 256, 0, stream>>>(Wv, wv_bf, W_N);
  cvt_f32_bf16<<<(W_N + 255) / 256, 256, 0, stream>>>(Wo, wo_bf, W_N);

  // 2) Q/K/V projections (WMMA GEMM + TDM staging, bf16 outputs)
  dim3 gg(Cc / 128, (Bc * Lc) / 128);
  gemm_bf16_kernel<true><<<gg, 256, 0, stream>>>(hs_bf, wq_bf, xc_bf, Bc * Lc, Cc, Cc);
  gemm_bf16_kernel<true><<<gg, 256, 0, stream>>>(hs_bf, wk_bf, xb_bf, Bc * Lc, Cc, Cc);
  gemm_bf16_kernel<true><<<gg, 256, 0, stream>>>(hs_bf, wv_bf, xa_bf, Bc * Lc, Cc, Cc);

  // 3) ilr gates
  ilr_kernel<<<(Bc * Lc * NHc) / 256, 256, 0, stream>>>(hs, ilr_w, ilr_b, ilr_f);

  // 4) sequential TTT scan: one block per (b, h), fast weights resident
  ttt_scan_kernel<<<Bc * NHc, 256, 0, stream>>>(xa_bf, xb_bf, xc_bf, ilr_f, W1g, W2g,
                                                y_bf);

  // 5) output projection -> f32 d_out
  gemm_bf16_kernel<false><<<gg, 256, 0, stream>>>(y_bf, wo_bf, d_out, Bc * Lc, Cc, Cc);
}
